// MoELayer_74113955660343
// MI455X (gfx1250) — matrix operand order repaired and checked by reading, not compiled, not run on hardware
//
#include <hip/hip_runtime.h>
#include <math.h>

// ---------------- problem constants ----------------
constexpr int T    = 8192;   // B*M*H*W tokens
constexpr int C    = 256;    // input features
constexpr int E    = 8;      // experts
constexpr int HID  = 512;    // MLP hidden
constexpr int COUT = 256;    // output features
constexpr int MT   = 64;     // token tile per block
constexpr int TILES_PER_E = T / MT;  // 128 (over-launch, early exit)

typedef __attribute__((ext_vector_type(16))) __bf16 v16bf;
typedef __attribute__((ext_vector_type(8)))  float  v8f;

union Frag { v16bf v; uint4 u[2]; };

// ---------------- kernel 1: weight transpose + f32->bf16 ----------------
// W1 [E,C,HID] -> W1t [E,HID,C]   (B-fragment needs contiguous K per column n)
// W2 [E,HID,COUT] -> W2t [E,COUT,HID]
__global__ __launch_bounds__(256) void prep_weights(const float* __restrict__ W1,
                                                    const float* __restrict__ W2,
                                                    __bf16* __restrict__ W1t,
                                                    __bf16* __restrict__ W2t) {
    int idx = blockIdx.x * 256 + threadIdx.x;      // < E*C*HID == E*HID*COUT
    {
        int e = idx / (C * HID), rem = idx % (C * HID);
        int c = rem / HID, h = rem % HID;
        W1t[((size_t)e * HID + h) * C + c] = (__bf16)W1[idx];
    }
    {
        int e = idx / (HID * COUT), rem = idx % (HID * COUT);
        int h = rem / COUT, o = rem % COUT;
        W2t[((size_t)e * COUT + o) * HID + h] = (__bf16)W2[idx];
    }
}

// ---------------- kernel 2: gate + routing + x->bf16 ----------------
// One wave (32 lanes) per token: coalesced 8-float chunks per lane.
__global__ __launch_bounds__(256) void gate_kernel(const float* __restrict__ x,
                                                   const float* __restrict__ Wg,
                                                   const float* __restrict__ bg,
                                                   const float* __restrict__ eb,
                                                   __bf16* __restrict__ xb,
                                                   int*   __restrict__ counts,
                                                   int*   __restrict__ toks,
                                                   float* __restrict__ wgts) {
    int lane = threadIdx.x & 31;
    int t    = (blockIdx.x * blockDim.x + threadIdx.x) >> 5;
    const float* xr = x + (size_t)t * C;

    float xv[8];
#pragma unroll
    for (int i = 0; i < 8; ++i) xv[i] = xr[lane * 8 + i];

    // bf16 copy of x (one 16B store per lane)
    __align__(16) __bf16 xb8[8];
#pragma unroll
    for (int i = 0; i < 8; ++i) xb8[i] = (__bf16)xv[i];
    *(uint4*)(xb + (size_t)t * C + lane * 8) = *(const uint4*)xb8;

    float acc[E] = {};
#pragma unroll
    for (int i = 0; i < 8; ++i) {
        int c = lane * 8 + i;
#pragma unroll
        for (int e = 0; e < E; ++e) acc[e] += xv[i] * Wg[c * E + e];
    }
#pragma unroll
    for (int e = 0; e < E; ++e) {
#pragma unroll
        for (int off = 16; off; off >>= 1) acc[e] += __shfl_down(acc[e], off, 32);
    }
    if (lane == 0) {
#pragma unroll
        for (int e = 0; e < E; ++e) acc[e] += bg[e] + eb[e];
        // top-2 (first-max tie-break matches lax.top_k)
        int i0 = 0; float v0 = acc[0];
#pragma unroll
        for (int e = 1; e < E; ++e) if (acc[e] > v0) { v0 = acc[e]; i0 = e; }
        int i1 = 0; float v1 = -INFINITY;
#pragma unroll
        for (int e = 0; e < E; ++e) if (e != i0 && acc[e] > v1) { v1 = acc[e]; i1 = e; }
        float p0 = 1.f / (1.f + expf(-v0));
        float p1 = 1.f / (1.f + expf(-v1));
        float s = p0 + p1; p0 /= s; p1 /= s;
        int q0 = atomicAdd(&counts[i0], 1);
        toks[i0 * T + q0] = t; wgts[i0 * T + q0] = p0;
        int q1 = atomicAdd(&counts[i1], 1);
        toks[i1 * T + q1] = t; wgts[i1 * T + q1] = p1;
    }
}

// ---------------- kernel 3: sparse expert MLP via WMMA bf16 ----------------
// Block = 8 waves, one 64-token tile of one expert.
// Each B fragment feeds 4 WMMAs (4 M-subtiles) -> 4x less weight traffic
// than a 16-token tile and much better load/WMMA overlap.
__global__ __launch_bounds__(256) void moe_mlp(const __bf16* __restrict__ xb,
                                               const __bf16* __restrict__ W1t,
                                               const __bf16* __restrict__ W2t,
                                               const float* __restrict__ b1,
                                               const float* __restrict__ b2,
                                               const int*   __restrict__ counts,
                                               const int*   __restrict__ toks,
                                               const float* __restrict__ wgts,
                                               float* __restrict__ out) {
    const int e    = blockIdx.x / TILES_PER_E;
    const int tile = blockIdx.x % TILES_PER_E;
    const int cnt  = counts[e];
    if (tile * MT >= cnt) return;            // uniform early exit

    __shared__ int   sTok[MT];
    __shared__ float sW[MT];
    // 64 KB buffer: first used as staged A (64 x 256 bf16 = 32 KB),
    // then (after GEMM1) reused as hdn (64 x 512 bf16 = 64 KB).
    __shared__ __align__(16) __bf16 smem[MT * HID];

    const int tid = threadIdx.x;
    if (tid < MT) {
        int i = tile * MT + tid;
        if (i < cnt) { sTok[tid] = toks[e * T + i]; sW[tid] = wgts[e * T + i]; }
        else         { sTok[tid] = 0;               sW[tid] = 0.f; }
    }
    __syncthreads();

    // ---- async-stage gathered token rows into LDS (CDNA5 async path) ----
    // 64 rows x 512B = 2048 x 16B chunks, 8 per thread; tracked by ASYNCcnt.
    {
        uint32_t ldsb = (uint32_t)(size_t)smem;
#pragma unroll
        for (int i = 0; i < 8; ++i) {
            int q    = i * 256 + tid;        // chunk index, 32 chunks per row
            int rowm = q >> 5;
            int c8   = (q & 31) * 8;         // element offset within row
            unsigned long long g =
                (unsigned long long)(xb + (size_t)sTok[rowm] * C + c8);
            uint32_t l = ldsb + (uint32_t)q * 16;
            asm volatile("global_load_async_to_lds_b128 %0, %1, off"
                         :: "v"(l), "v"(g) : "memory");
        }
        asm volatile("s_wait_asynccnt 0x0" ::: "memory");
    }
    __syncthreads();

    const int lane  = threadIdx.x & 31;
    const int wave  = threadIdx.x >> 5;
    const int row   = lane & 15;          // N (or M) position within a 16-tile
    const int khalf = lane >> 4;          // 0: lanes 0-15, 1: lanes 16-31

    const __bf16* w1e = W1t + (size_t)e * HID * C;
    const __bf16* w2e = W2t + (size_t)e * COUT * HID;

    // ---- GEMM1: [64,C] x [C,HID]; wave owns 4 N-tiles, 4 M-subtiles ----
    v8f acc1[16];
#pragma unroll
    for (int j = 0; j < 16; ++j) acc1[j] = (v8f){0.f,0.f,0.f,0.f,0.f,0.f,0.f,0.f};

#pragma unroll
    for (int ks = 0; ks < C / 32; ++ks) {
        int ka = ks * 32 + khalf * 8;                 // A chunks {ka.., ka+16..}
        Frag a[4];
#pragma unroll
        for (int ms = 0; ms < 4; ++ms) {              // A from LDS (ds_load_b128)
            const __bf16* ap = smem + ((ms * 16 + row) * C + ka);
            a[ms].u[0] = *(const uint4*)ap;
            a[ms].u[1] = *(const uint4*)(ap + 16);
        }
        int kb = ks * 32 + khalf * 8;                 // B chunks {kb.., kb+16..}
#pragma unroll
        for (int j = 0; j < 4; ++j) {
            int n = (wave * 4 + j) * 16 + row;
            const __bf16* bp = w1e + (size_t)n * C + kb;
            Frag b;
            b.u[0] = *(const uint4*)bp;
            b.u[1] = *(const uint4*)(bp + 16);
#pragma unroll
            for (int ms = 0; ms < 4; ++ms) {
                acc1[j * 4 + ms] = __builtin_amdgcn_wmma_f32_16x16x32_bf16(
                    false, a[ms].v, false, b.v, (short)0, acc1[j * 4 + ms], false, false);
            }
        }
    }
    __syncthreads();   // all A reads done before smem is rewritten as hdn

    // ---- bias + exact GELU, stage hdn to LDS as bf16 [64, HID] ----
#pragma unroll
    for (int j = 0; j < 4; ++j) {
        int n = (wave * 4 + j) * 16 + row;
        float bias = b1[e * HID + n];
#pragma unroll
        for (int ms = 0; ms < 4; ++ms) {
#pragma unroll
            for (int r = 0; r < 8; ++r) {
                float v = acc1[j * 4 + ms][r] + bias;
                v = 0.5f * v * (1.0f + erff(v * 0.70710678118654752f));
                int m = ms * 16 + r + khalf * 8;
                smem[m * HID + n] = (__bf16)v;
            }
        }
    }
    __syncthreads();

    // ---- GEMM2: [64,HID] x [HID,COUT]; wave owns 2 N-tiles, 4 M-subtiles ----
    v8f acc2[8];
#pragma unroll
    for (int j = 0; j < 8; ++j) acc2[j] = (v8f){0.f,0.f,0.f,0.f,0.f,0.f,0.f,0.f};

#pragma unroll
    for (int ks = 0; ks < HID / 32; ++ks) {
        int ka = ks * 32 + khalf * 8;
        Frag a[4];
#pragma unroll
        for (int ms = 0; ms < 4; ++ms) {              // A = hdn from LDS
            const __bf16* ap = smem + ((ms * 16 + row) * HID + ka);
            a[ms].u[0] = *(const uint4*)ap;
            a[ms].u[1] = *(const uint4*)(ap + 16);
        }
        int kb = ks * 32 + khalf * 8;
#pragma unroll
        for (int j = 0; j < 2; ++j) {
            int n = (wave * 2 + j) * 16 + row;
            const __bf16* bp = w2e + (size_t)n * HID + kb;
            Frag b;
            b.u[0] = *(const uint4*)bp;
            b.u[1] = *(const uint4*)(bp + 16);
#pragma unroll
            for (int ms = 0; ms < 4; ++ms) {
                acc2[j * 4 + ms] = __builtin_amdgcn_wmma_f32_16x16x32_bf16(
                    false, a[ms].v, false, b.v, (short)0, acc2[j * 4 + ms], false, false);
            }
        }
    }

    // ---- epilogue: +b2, scale by routing weight, scatter-add ----
#pragma unroll
    for (int j = 0; j < 2; ++j) {
        int n = (wave * 2 + j) * 16 + row;
        float bias = b2[e * COUT + n];
#pragma unroll
        for (int ms = 0; ms < 4; ++ms) {
#pragma unroll
            for (int r = 0; r < 8; ++r) {
                int m = ms * 16 + r + khalf * 8;
                float y = (acc2[j * 4 + ms][r] + bias) * sW[m];
                atomicAdd(&out[(size_t)sTok[m] * COUT + n], y);
            }
        }
    }
}

// ---------------- host launcher ----------------
extern "C" void kernel_launch(void* const* d_in, const int* in_sizes, int n_in,
                              void* d_out, int out_size, void* d_ws, size_t ws_size,
                              hipStream_t stream) {
    const float* x   = (const float*)d_in[0];
    const float* Wg  = (const float*)d_in[1];
    const float* bg  = (const float*)d_in[2];
    const float* eb  = (const float*)d_in[3];
    const float* W1  = (const float*)d_in[4];
    const float* b1  = (const float*)d_in[5];
    const float* W2  = (const float*)d_in[6];
    const float* b2  = (const float*)d_in[7];
    float* out = (float*)d_out;

    // workspace layout (256B aligned)
    char* ws = (char*)d_ws;
    size_t off = 0;
    __bf16* xb  = (__bf16*)(ws + off); off += (size_t)T * C * 2;          // 4 MB
    __bf16* W1t = (__bf16*)(ws + off); off += (size_t)E * HID * C * 2;    // 2 MB
    __bf16* W2t = (__bf16*)(ws + off); off += (size_t)E * COUT * HID * 2; // 2 MB
    int*    counts = (int*)(ws + off); off += 256;
    int*    toks   = (int*)(ws + off); off += (size_t)E * T * 4;          // 256 KB
    float*  wgts   = (float*)(ws + off); off += (size_t)E * T * 4;        // 256 KB
    (void)ws_size; (void)in_sizes; (void)n_in;

    hipMemsetAsync(counts, 0, 256, stream);
    hipMemsetAsync(out, 0, (size_t)out_size * sizeof(float), stream);

    // weights -> bf16 transposed
    prep_weights<<<(E * C * HID) / 256, 256, 0, stream>>>(W1, W2, W1t, W2t);
    // gating + routing lists + x -> bf16 (wave per token)
    gate_kernel<<<(T * 32) / 256, 256, 0, stream>>>(x, Wg, bg, eb, xb, counts, toks, wgts);
    // sparse expert MLP (over-launched grid, early exit past per-expert counts)
    moe_mlp<<<E * TILES_PER_E, 256, 0, stream>>>(xb, W1t, W2t, b1, b2, counts, toks, wgts, out);
}
